// SepConvLayer_75608604279010
// MI455X (gfx1250) — matrix operand order repaired and checked by reading, not compiled, not run on hardware
//
#include <hip/hip_runtime.h>

// ---------------------------------------------------------------------------
// Problem constants (match the reference)
// ---------------------------------------------------------------------------
#define BB 8
#define CC 64
#define PP 64
#define HH 256
#define WW 256
#define NPIX (BB * HH * WW)          // 524288 pixels, also the output size
#define BN_EPS 1e-5f

// chain tiling
#define STEPS 16                     // chain steps fused per kernel
#define TILE  32                     // output tile edge
#define HALO  16                     // = STEPS
#define IN    64                     // TILE + 2*HALO
#define PADW  65                     // LDS row stride (bank-conflict pad)

typedef __attribute__((ext_vector_type(16))) _Float16 v16h;
typedef __attribute__((ext_vector_type(8)))  float    v8f;

// ---------------------------------------------------------------------------
// Prep: fold BatchNorm into scale/bias
// ---------------------------------------------------------------------------
__global__ void bn_prep_kernel(const float* __restrict__ gamma,
                               const float* __restrict__ beta,
                               const float* __restrict__ mean,
                               const float* __restrict__ var,
                               float* __restrict__ sc,
                               float* __restrict__ bi) {
    int p = threadIdx.x;
    if (p < PP) {
        float s = gamma[p] * rsqrtf(var[p] + BN_EPS);
        sc[p] = s;
        bi[p] = beta[p] - mean[p] * s;
    }
}

// ---------------------------------------------------------------------------
// Depthwise chain: 16 fused 3x3+ReLU6 steps per kernel, LDS-resident.
//
// 64x64 input tile (32x32 output + 16 halo) ping-pongs between two LDS
// buffers; after step s, rings s..63-s are valid, so the central 32x32 is
// valid at every step and is streamed out to its ys plane (f16).
// Image-edge zero padding is re-imposed every step.
// ---------------------------------------------------------------------------
template <bool FIRST>
__global__ void __launch_bounds__(256)
dw_chain_kernel(const void* __restrict__ srcv,    // FIRST: x f32 [B,C,H,W]
                                                  // else : ys plane c0-1 f16
                _Float16* __restrict__ ysbuf,     // [C][NPIX] f16 planes
                const float* __restrict__ dw_w,   // [C,3,3] f32
                int c0) {                         // first step index
    __shared__ float buf[2][IN][PADW];            // 2 x 16.6 KB

    const int t   = threadIdx.x;
    const int tx  = blockIdx.x & 7;
    const int ty  = (blockIdx.x >> 3) & 7;
    const int b   = blockIdx.x >> 6;
    const int gx0 = tx * TILE - HALO;
    const int gy0 = ty * TILE - HALO;

    // ---- load 64x64 input tile (zero outside the image) --------------------
    for (int i = t; i < IN * IN; i += 256) {
        const int ly = i >> 6, lx = i & 63;
        const int gy = gy0 + ly, gx = gx0 + lx;
        float v = 0.f;
        if ((unsigned)gy < (unsigned)HH && (unsigned)gx < (unsigned)WW) {
            const int n = (gy << 8) + gx;
            if (FIRST)
                v = ((const float*)srcv)[b * (CC << 16) + n];     // x[:,0]
            else
                v = (float)((const _Float16*)srcv)[(b << 16) + n];
        }
        buf[0][ly][lx] = v;
    }
    __syncthreads();

    int cur = 0;
    for (int s = 0; s < STEPS; ++s) {
        const float* w = dw_w + (size_t)(c0 + s) * 9;   // uniform -> SGPRs
        const float w0 = w[0], w1 = w[1], w2 = w[2];
        const float w3 = w[3], w4 = w[4], w5 = w[5];
        const float w6 = w[6], w7 = w[7], w8 = w[8];

        // compute interior rows [1,63) x cols [1,63)
        for (int i = t; i < 62 * 64; i += 256) {
            const int ly = (i >> 6) + 1;
            const int lx = i & 63;
            if (lx == 0 || lx == 63) continue;
            float acc = w0 * buf[cur][ly - 1][lx - 1] +
                        w1 * buf[cur][ly - 1][lx    ] +
                        w2 * buf[cur][ly - 1][lx + 1] +
                        w3 * buf[cur][ly    ][lx - 1] +
                        w4 * buf[cur][ly    ][lx    ] +
                        w5 * buf[cur][ly    ][lx + 1] +
                        w6 * buf[cur][ly + 1][lx - 1] +
                        w7 * buf[cur][ly + 1][lx    ] +
                        w8 * buf[cur][ly + 1][lx + 1];
            acc = fminf(fmaxf(acc, 0.f), 6.f);          // ReLU6
            // re-impose image-edge zero padding
            const int gy = gy0 + ly, gx = gx0 + lx;
            if ((unsigned)gy >= (unsigned)HH || (unsigned)gx >= (unsigned)WW)
                acc = 0.f;
            buf[cur ^ 1][ly][lx] = acc;
        }
        __syncthreads();

        // stream this step's central 32x32 to its ys plane (64B row segments)
        _Float16* dst = ysbuf + (size_t)(c0 + s) * NPIX + (b << 16);
        for (int i = t; i < TILE * TILE; i += 256) {
            const int ly = i >> 5, lx = i & 31;
            const int gy = gy0 + HALO + ly, gx = gx0 + HALO + lx;
            dst[(gy << 8) + gx] = (_Float16)buf[cur ^ 1][HALO + ly][HALO + lx];
        }
        cur ^= 1;
        // no extra sync needed: next compute writes the other buffer only
    }
}

// ---------------------------------------------------------------------------
// Fused pointwise GEMM (64x64 over pixels, WMMA f16->f32) + BN + ReLU6 +
// conv_w reduction P->1.
//   A (pw_w 16x32 f16): lane L: M=L%16; elem e:
//       K = e + 16*(L/16)
//   B (ys 32x16 f16):   lane L: N=L%16; elem e: K = e + 16*(L/16)
//   D (16x16 f32):      reg r:  M = r + 8*(L/16), N = L%16
// ---------------------------------------------------------------------------
#define GROUPS_PER_WAVE 8   // 128 pixels per wave

__global__ void __launch_bounds__(256)
pw_gemm_kernel(const _Float16* __restrict__ ys,   // [C][NPIX] f16 planes
               const float* __restrict__ pw_w,    // [P][C] f32
               const float* __restrict__ sc,      // [P]
               const float* __restrict__ bi,      // [P]
               const float* __restrict__ cw,      // conv_w [1][P]
               float* __restrict__ out) {         // [NPIX]
    const int lane = threadIdx.x & 31;
    const int g    = lane >> 4;
    const int ln   = lane & 15;
    const int wave = blockIdx.x * (blockDim.x >> 5) + (threadIdx.x >> 5);
    const int n_base = wave * (16 * GROUPS_PER_WAVE);

    // A fragments: 4 P-tiles x 2 K-tiles, f32 -> f16 (loaded once, L2 hits)
    v16h a[4][2];
#pragma unroll
    for (int pt = 0; pt < 4; ++pt) {
#pragma unroll
        for (int kt = 0; kt < 2; ++kt) {
#pragma unroll
            for (int e = 0; e < 16; ++e) {
                const int k = e + 16 * g + 32 * kt;
                a[pt][kt][e] = (_Float16)pw_w[(16 * pt + ln) * CC + k];
            }
        }
    }

    // epilogue constants for this lane's 32 P values
    float scv[32], biv[32], cwv[32];
#pragma unroll
    for (int pt = 0; pt < 4; ++pt) {
#pragma unroll
        for (int r = 0; r < 8; ++r) {
            const int p = 16 * pt + 8 * g + r;
            scv[pt * 8 + r] = sc[p];
            biv[pt * 8 + r] = bi[p];
            cwv[pt * 8 + r] = cw[p];
        }
    }

    for (int grp = 0; grp < GROUPS_PER_WAVE; ++grp) {
        const int n0 = n_base + grp * 16;

        // B fragments: coalesced (16 consecutive halves per lane half and element index)
        v16h b0, b1;
#pragma unroll
        for (int e = 0; e < 16; ++e) {
            const int c = e + 16 * g;
            b0[e] = ys[c * NPIX + n0 + ln];
            b1[e] = ys[(c + 32) * NPIX + n0 + ln];
        }

        // 8 WMMAs: D[pt] = A[pt] * B  (K=64 in two steps of 32)
        v8f d[4];
#pragma unroll
        for (int pt = 0; pt < 4; ++pt) {
            v8f acc = {};
            acc = __builtin_amdgcn_wmma_f32_16x16x32_f16(
                false, a[pt][0], false, b0, (short)0, acc, false, false);
            acc = __builtin_amdgcn_wmma_f32_16x16x32_f16(
                false, a[pt][1], false, b1, (short)0, acc, false, false);
            d[pt] = acc;
        }

        // BN + ReLU6 + conv_w reduction over this lane's 32 P values
        float partial = 0.f;
#pragma unroll
        for (int pt = 0; pt < 4; ++pt) {
#pragma unroll
            for (int r = 0; r < 8; ++r) {
                const int i = pt * 8 + r;
                float hv = d[pt][r] * scv[i] + biv[i];
                hv = fminf(fmaxf(hv, 0.f), 6.f);
                partial += cwv[i] * hv;
            }
        }
        partial += __shfl_xor(partial, 16, 32);   // fold the two M-halves
        if (g == 0)
            out[n0 + ln] = partial;
    }
}

// ---------------------------------------------------------------------------
// Launch: prep + 4 fused-chain kernels (16 steps each, LDS-resident,
// block-local sync) + 1 WMMA GEMM kernel.  6 launches total.
// Workspace: [sc 64 f32][bi 64 f32][ys: 64 planes x NPIX f16 = 64 MB]
// ---------------------------------------------------------------------------
extern "C" void kernel_launch(void* const* d_in, const int* in_sizes, int n_in,
                              void* d_out, int out_size, void* d_ws, size_t ws_size,
                              hipStream_t stream) {
    (void)in_sizes; (void)n_in; (void)out_size; (void)ws_size;

    const float* x      = (const float*)d_in[0];   // [B,C,H,W]
    const float* dw_w   = (const float*)d_in[1];   // [C,3,3]
    const float* pw_w   = (const float*)d_in[2];   // [P,C]
    const float* gamma  = (const float*)d_in[3];
    const float* beta   = (const float*)d_in[4];
    const float* mean   = (const float*)d_in[5];
    const float* var    = (const float*)d_in[6];
    const float* conv_w = (const float*)d_in[7];   // [1,P]

    float*    sc    = (float*)d_ws;
    float*    bi    = sc + PP;
    _Float16* ysbuf = (_Float16*)(sc + 2 * PP);    // 64 planes of NPIX f16

    bn_prep_kernel<<<1, 64, 0, stream>>>(gamma, beta, mean, var, sc, bi);

    // 8x8 tiles per image x 8 batches = 512 blocks per chain kernel
    const dim3 chain_grid(8 * 8 * BB);
    dw_chain_kernel<true><<<chain_grid, 256, 0, stream>>>(
        (const void*)x, ysbuf, dw_w, 0);
    for (int c0 = STEPS; c0 < CC; c0 += STEPS) {
        dw_chain_kernel<false><<<chain_grid, 256, 0, stream>>>(
            (const void*)(ysbuf + (size_t)(c0 - 1) * NPIX), ysbuf, dw_w, c0);
    }

    // 16*GROUPS_PER_WAVE px/wave, 8 waves/block -> 1024 px/block, 512 blocks
    const dim3 gemm_grid(NPIX / (16 * GROUPS_PER_WAVE * 8));
    pw_gemm_kernel<<<gemm_grid, 256, 0, stream>>>(
        ysbuf, pw_w, sc, bi, conv_w, (float*)d_out);
}
